// SemanticAttention_54692113547383
// MI455X (gfx1250) — matrix operand order repaired and checked by reading, not compiled, not run on hardware
//
#include <hip/hip_runtime.h>
#include <hip/hip_bf16.h>
#include <cstdint>

#define BB 4
#define LL 2048
#define DD 1024
#define HH 16
#define DKK 64

typedef __attribute__((ext_vector_type(16))) __bf16 v16bf;
typedef __attribute__((ext_vector_type(8)))  float  v8f;
typedef __attribute__((ext_vector_type(4)))  unsigned int u32x4;
typedef __attribute__((ext_vector_type(8)))  unsigned int u32x8;

__device__ __forceinline__ __bf16 f2bf(float f) { return (__bf16)f; }

// --- Async DMA of one 16B chunk: global -> LDS, tracked on ASYNCcnt ---------
__device__ __forceinline__ void async_copy16(uint32_t lds_addr, const void* gptr) {
    asm volatile("global_load_async_to_lds_b128 %0, %1, off"
                 :: "v"(lds_addr), "v"((uint64_t)(uintptr_t)gptr)
                 : "memory");
}
__device__ __forceinline__ void async_wait0() {
    asm volatile("s_wait_asynccnt 0x0" ::: "memory");
}

// --- Tensor Data Mover: 2D tile (rows x 32 f32, row stride = stride_elts) ---
// D# built per CDNA5 ISA §8.3/§8.4. Dense LDS fill (no padding) matches a
// row-major [rows][32] f32 LDS tile. Groups 2/3 zero (2D tensor).
__device__ __forceinline__ void tdm_load_tile_2d(uint32_t lds_addr, const void* gptr,
                                                 uint32_t rows, uint32_t stride_elts) {
    const uint64_t ga = (uint64_t)(uintptr_t)gptr;
    u32x4 g0;
    g0[0] = 1u;                                            // count=1 (valid, user mode)
    g0[1] = lds_addr;                                      // lds_addr [63:32]
    g0[2] = (uint32_t)ga;                                  // global_addr [95:64]
    g0[3] = (uint32_t)((ga >> 32) & 0x01FFFFFFu)           // global_addr [120:96]
          | (2u << 30);                                    // type=2 ("image") [127:126]
    u32x8 g1;
    g1[0] = (2u << 16);          // workgroup_mask=0, data_size=2 (4B), no flags
    g1[1] = (32u << 16);         // atomic_barrier_addr=0; tensor_dim0[15:0]=32
    g1[2] = (rows << 16);        // tensor_dim0[31:16]=0; tensor_dim1[15:0]=rows
    g1[3] = (32u << 16);         // tensor_dim1[31:16]=0; tile_dim0=32
    g1[4] = rows;                // tile_dim1=rows; tile_dim2=0
    g1[5] = stride_elts;         // tensor_dim0_stride[31:0] (f32 elements)
    g1[6] = 0u;                  // dim0_stride[47:32]=0; dim1_stride[15:0]=0
    g1[7] = 0u;
    u32x4 gz = {0u, 0u, 0u, 0u};
    asm volatile("tensor_load_to_lds %0, %1, %2, %3"
                 :: "s"(g0), "s"(g1), "s"(gz), "s"(gz)
                 : "memory");
}

// ---------------------------------------------------------------------------
// C[M,N] = A[M,K] @ W[N,K]^T + bias[N] (+ addend[M,N])
// Block tile 128x128, 8 waves (2x4), wave tile 64x32 = 4x2 WMMA tiles.
// f32 tiles double-buffered in LDS via the Tensor Data Mover (wave 0 issues
// one tensor_load_to_lds per tile, TENSORcnt + barrier sync); f32->bf16
// conversion fused into fragment build. grid: (M/128, N/128), block 256.
// ---------------------------------------------------------------------------
template <bool HasAdd>
__global__ __launch_bounds__(256)
void gemm_bt_kernel(const float* __restrict__ A, const float* __restrict__ W,
                    const float* __restrict__ bias, const float* __restrict__ addend,
                    float* __restrict__ C, int N, int K) {
    __shared__ __attribute__((aligned(16))) float sA[2][128 * 32];
    __shared__ __attribute__((aligned(16))) float sB[2][128 * 32];

    const int t    = threadIdx.x;
    const int wave = t >> 5;
    const int lane = t & 31;
    const int wm   = wave >> 2;     // 0..1 : 64-row strip
    const int wn   = wave & 3;      // 0..3 : 32-col strip
    const int row  = lane & 15;
    const int hi   = lane >> 4;

    const int m0 = blockIdx.x * 128;
    const int n0 = blockIdx.y * 128;
    const int KT = K >> 5;          // #32-wide K steps

    auto stage = [&](int buf, int kt) {   // wave 0 only: 2 TDM tile loads
        const int k0 = kt << 5;
        tdm_load_tile_2d((uint32_t)(uintptr_t)&sA[buf][0],
                         A + (size_t)m0 * K + k0, 128u, (uint32_t)K);
        tdm_load_tile_2d((uint32_t)(uintptr_t)&sB[buf][0],
                         W + (size_t)n0 * K + k0, 128u, (uint32_t)K);
    };

    v8f acc[4][2] = {};
    if (wave == 0) stage(0, 0);

    for (int kt = 0; kt < KT; ++kt) {
        const int buf = kt & 1;
        if (wave == 0) __builtin_amdgcn_s_wait_tensorcnt(0);  // buf's tiles landed
        __syncthreads();                                      // publish to all waves
        if (kt + 1 < KT && wave == 0) stage(buf ^ 1, kt + 1); // overlap DMA w/ compute

        // B fragments (2 n-tiles), reused by all 4 m-tiles
        v16bf bfr[2];
#pragma unroll
        for (int nt = 0; nt < 2; ++nt) {
            const float* src = &sB[buf][(wn * 32 + nt * 16 + row) * 32 + hi * 16];
#pragma unroll
            for (int i = 0; i < 16; ++i) bfr[nt][i] = f2bf(src[i]);
        }
#pragma unroll
        for (int mt = 0; mt < 4; ++mt) {
            const float* ar = &sA[buf][(wm * 64 + mt * 16 + row) * 32];
            v16bf af;
#pragma unroll
            for (int i = 0; i < 8; ++i) {
                af[i]     = f2bf(ar[hi * 16 + i]);
                af[8 + i] = f2bf(ar[hi * 16 + 8 + i]);
            }
#pragma unroll
            for (int nt = 0; nt < 2; ++nt)
                acc[mt][nt] = __builtin_amdgcn_wmma_f32_16x16x32_bf16(
                    false, af, false, bfr[nt], (short)0, acc[mt][nt], false, false);
        }
    }

#pragma unroll
    for (int nt = 0; nt < 2; ++nt) {
        const int n  = n0 + wn * 32 + nt * 16 + row;
        const float bv = bias[n];
#pragma unroll
        for (int mt = 0; mt < 4; ++mt) {
            const int m = m0 + wm * 64 + mt * 16 + hi * 8;
#pragma unroll
            for (int r = 0; r < 8; ++r) {
                const size_t idx = (size_t)(m + r) * N + n;
                float val = acc[mt][nt][r] + bv;
                if (HasAdd) val += addend[idx];
                C[idx] = val;
            }
        }
    }
}

// ---------------------------------------------------------------------------
// Scores: S[bh,m,n] = (Qh[m,:] . Kh[n,:]) / 8, masked. The 16x64 Q tile is
// shared by all 8 waves -> staged once per block via async-to-LDS; each wave
// then amortizes its A fragment over 4 n-tiles.
// grid: (L/16, 4, B*H), block 256.
// ---------------------------------------------------------------------------
__global__ __launch_bounds__(256)
void score_kernel(const float* __restrict__ Qp, const float* __restrict__ Kp,
                  const int* __restrict__ mask, float* __restrict__ attn) {
    __shared__ __attribute__((aligned(16))) float sQ[16 * 64];

    const int t    = threadIdx.x;
    const int wave = t >> 5;
    const int lane = t & 31;
    const int row  = lane & 15;
    const int hi   = lane >> 4;

    const int bh = blockIdx.z;
    const int b  = bh >> 4;
    const int h  = bh & 15;
    const int tm  = blockIdx.x;
    const int tn0 = (blockIdx.y * 8 + wave) * 4;   // 4 n-tiles per wave
    const int hbase = h * DKK;

    // Stage Q tile: 16 rows x 64 f32 = 256 x 16B chunks, one per thread.
    {
        const int r  = t >> 4;          // 0..15
        const int c4 = (t & 15) * 4;    // 0..60
        async_copy16((uint32_t)(uintptr_t)&sQ[r * 64 + c4],
                     Qp + ((size_t)b * LL + tm * 16 + r) * DD + hbase + c4);
    }
    async_wait0();
    __syncthreads();

    v8f acc[4] = {};
#pragma unroll
    for (int k0 = 0; k0 < DKK; k0 += 32) {
        v16bf af;
        const float* qr = &sQ[row * 64 + k0];
#pragma unroll
        for (int i = 0; i < 8; ++i) {
            af[i]     = f2bf(qr[hi * 16 + i]);
            af[8 + i] = f2bf(qr[hi * 16 + 8 + i]);
        }
#pragma unroll
        for (int j = 0; j < 4; ++j) {
            const float* krow =
                Kp + ((size_t)b * LL + (tn0 + j) * 16 + row) * DD + hbase + k0 + hi * 16;
            v16bf bf;
#pragma unroll
            for (int i = 0; i < 16; ++i) bf[i] = f2bf(krow[i]);
            acc[j] = __builtin_amdgcn_wmma_f32_16x16x32_bf16(
                false, af, false, bf, (short)0, acc[j], false, false);
        }
    }

    const int m0 = tm * 16 + hi * 8;
#pragma unroll
    for (int j = 0; j < 4; ++j) {
        const int n = (tn0 + j) * 16 + row;
        const bool masked = (mask[(size_t)b * LL + n] == 0);
        float* out = attn + ((size_t)bh * LL + m0) * LL + n;
#pragma unroll
        for (int r = 0; r < 8; ++r) {
            out[(size_t)r * LL] = masked ? -1.0e9f : acc[j][r] * 0.125f;
        }
    }
}

// ---------------------------------------------------------------------------
// In-place row softmax over L=2048. grid: B*H*L rows, block 256.
// ---------------------------------------------------------------------------
__global__ __launch_bounds__(256)
void softmax_kernel(float* __restrict__ attn) {
    __shared__ float red[256];
    float* p = attn + (size_t)blockIdx.x * LL;
    const int t = threadIdx.x;

    float mx = -3.4e38f;
    for (int i = t; i < LL; i += 256) mx = fmaxf(mx, p[i]);
    red[t] = mx; __syncthreads();
    for (int s = 128; s > 0; s >>= 1) {
        if (t < s) red[t] = fmaxf(red[t], red[t + s]);
        __syncthreads();
    }
    mx = red[0]; __syncthreads();

    float sum = 0.0f;
    for (int i = t; i < LL; i += 256) {
        float e = __expf(p[i] - mx);
        p[i] = e;
        sum += e;
    }
    red[t] = sum; __syncthreads();
    for (int s = 128; s > 0; s >>= 1) {
        if (t < s) red[t] += red[t + s];
        __syncthreads();
    }
    const float inv = 1.0f / red[0];
    for (int i = t; i < LL; i += 256) p[i] *= inv;
}

// ---------------------------------------------------------------------------
// O[b, m, h*DK + n] = sum_k attn[bh,m,k] * Vp[b,k,h*DK+n]
// Block 128 (4 waves = 4 n-tiles over DK=64); each wave does 2 m-tiles,
// sharing the B(V) fragment. V fragment loads are 64B-coalesced per
// half-wave; V slices (512KB/head) live in L2; no per-block duplication.
// grid: (L/32, B*H).
// ---------------------------------------------------------------------------
__global__ __launch_bounds__(128)
void av_kernel(const float* __restrict__ attn, const float* __restrict__ Vp,
               float* __restrict__ O) {
    const int wave = threadIdx.x >> 5;
    const int lane = threadIdx.x & 31;
    const int row  = lane & 15;
    const int hi   = lane >> 4;

    const int bh = blockIdx.y;
    const int b  = bh >> 4;
    const int h  = bh & 15;
    const int tm0 = blockIdx.x * 2;   // 2 m-tiles per wave
    const int tn  = wave;             // 0..3

    const float* arow0 = attn + ((size_t)bh * LL + tm0 * 16 + row) * LL;
    const float* arow1 = arow0 + (size_t)16 * LL;
    const float* vcol  = Vp + (size_t)b * LL * DD + h * DKK + tn * 16 + row;

    v8f acc[2] = {};
    for (int k0 = 0; k0 < LL; k0 += 32) {
        v16bf bf;
#pragma unroll
        for (int i = 0; i < 16; ++i)
            bf[i] = f2bf(vcol[(size_t)(k0 + hi * 16 + i) * DD]);

        const float* arows[2] = {arow0, arow1};
#pragma unroll
        for (int mt = 0; mt < 2; ++mt) {
            v16bf af;
#pragma unroll
            for (int i = 0; i < 8; ++i) {
                af[i]     = f2bf(arows[mt][k0 + hi * 16 + i]);
                af[8 + i] = f2bf(arows[mt][k0 + hi * 16 + 8 + i]);
            }
            acc[mt] = __builtin_amdgcn_wmma_f32_16x16x32_bf16(
                false, af, false, bf, (short)0, acc[mt], false, false);
        }
    }

    const int n = h * DKK + tn * 16 + row;
#pragma unroll
    for (int mt = 0; mt < 2; ++mt) {
        const int m0 = (tm0 + mt) * 16 + hi * 8;
#pragma unroll
        for (int r = 0; r < 8; ++r)
            O[((size_t)b * LL + m0 + r) * DD + n] = acc[mt][r];
    }
}

// ---------------------------------------------------------------------------
extern "C" void kernel_launch(void* const* d_in, const int* in_sizes, int n_in,
                              void* d_out, int out_size, void* d_ws, size_t ws_size,
                              hipStream_t stream) {
    const float* q    = (const float*)d_in[0];
    const float* k    = (const float*)d_in[1];
    const float* v    = (const float*)d_in[2];
    const float* semf = (const float*)d_in[3];
    const int*   mask = (const int*)  d_in[4];
    const float* w_q  = (const float*)d_in[5];
    const float* b_q  = (const float*)d_in[6];
    const float* w_k  = (const float*)d_in[7];
    const float* b_k  = (const float*)d_in[8];
    const float* w_v  = (const float*)d_in[9];
    const float* b_v  = (const float*)d_in[10];
    const float* w_s  = (const float*)d_in[11];
    const float* b_s  = (const float*)d_in[12];
    const float* w_o  = (const float*)d_in[13];
    const float* b_o  = (const float*)d_in[14];

    float* out  = (float*)d_out;                  // [B, L, D]
    float* attn = out + (size_t)BB * LL * DD;     // [B, H, L, L]

    const size_t BLD = (size_t)BB * LL * DD;
    float* sem = (float*)d_ws;
    float* qp  = sem + BLD;
    float* kp  = qp  + BLD;
    float* vp  = kp  + BLD;
    float* o   = vp  + BLD;

    const dim3 blk256(256);
    const dim3 gproj(BB * LL / 128, DD / 128);    // (64, 8)

    // Projections (sem first; k/v fuse the +sem addend)
    gemm_bt_kernel<false><<<gproj, blk256, 0, stream>>>(semf, w_s, b_s, nullptr, sem, DD, 1024);
    gemm_bt_kernel<false><<<gproj, blk256, 0, stream>>>(q,    w_q, b_q, nullptr, qp,  DD, DD);
    gemm_bt_kernel<true ><<<gproj, blk256, 0, stream>>>(k,    w_k, b_k, sem,     kp,  DD, DD);
    gemm_bt_kernel<true ><<<gproj, blk256, 0, stream>>>(v,    w_v, b_v, sem,     vp,  DD, DD);

    // Attention scores + mask -> logits in attn region of d_out
    score_kernel<<<dim3(LL / 16, 4, BB * HH), blk256, 0, stream>>>(qp, kp, mask, attn);

    // Softmax in place (attn becomes the final probability output)
    softmax_kernel<<<dim3(BB * HH * LL), blk256, 0, stream>>>(attn);

    // O = attn @ V, written directly in merged (B, L, D) layout
    av_kernel<<<dim3(LL / 32, BB * HH), dim3(128), 0, stream>>>(attn, vp, o);

    // Output projection
    gemm_bt_kernel<false><<<gproj, blk256, 0, stream>>>(o, w_o, b_o, nullptr, out, DD, DD);
}
